// LstmMemoryPredict_26809185861927
// MI455X (gfx1250) — matrix operand order repaired and checked by reading, not compiled, not run on hardware
//
#include <hip/hip_runtime.h>
#include <hip/hip_bf16.h>

#define T_LEN 2048
#define B_SZ  256
#define HID   32
#define G4H   128   // 4*HID

typedef _Float16 h8  __attribute__((ext_vector_type(8)));
typedef _Float16 h4  __attribute__((ext_vector_type(4)));
typedef _Float16 v16h __attribute__((ext_vector_type(16)));
typedef float    v8f  __attribute__((ext_vector_type(8)));

// ---- fast transcendentals: keep the recurrence critical path on TRANS32 ops --
__device__ __forceinline__ float fast_tanh(float x) {
#if __has_builtin(__builtin_amdgcn_tanhf)
  return __builtin_amdgcn_tanhf(x);          // v_tanh_f32 (CDNA5 TRANS op)
#else
  // tanh(x) = 1 - 2/(exp2(x*2/ln2)+1): v_exp_f32 + v_rcp_f32, no IEEE divide
  float e = __builtin_amdgcn_exp2f(x * 2.8853900817779268f);
  return 1.0f - 2.0f * __builtin_amdgcn_rcpf(e + 1.0f);
#endif
}

// ---- A-operand builders for v_wmma_f32_16x16x32_f16 -------------------------
// Per ISA 16-bit A 16x32 layout: lane holds K = kbase..kbase+7 (elems 0..7) and
// K = kbase+16..kbase+23 (elems 8..15), kbase = 8*(lane>>4). `p` already points
// at this lane's kbase within the 32-wide K window of row M = lane&15.
__device__ __forceinline__ v16h a_from_f16(const _Float16* p) {
  h8 lo = *(const h8*)p;
  h8 hi = *(const h8*)(p + 16);
  return __builtin_shufflevector(lo, hi, 0,1,2,3,4,5,6,7,8,9,10,11,12,13,14,15);
}
__device__ __forceinline__ v16h a_from_f32(const float* p) {
  v16h a;
#pragma unroll
  for (int i = 0; i < 8; ++i) { a[i] = (_Float16)p[i]; a[i+8] = (_Float16)p[16+i]; }
  return a;
}

// ---- weight prep: f32 -> f16, bias = b_ih + b_hh ----------------------------
// wihf layout: layer0 [128x64] at 0, layers 1..4 [128x32] at 8192+(l-1)*4096.
// whhf layout: layer l [128x32] at l*4096.
__global__ void prep_kernel(const float* __restrict__ wih0, const float* __restrict__ wihr,
                            const float* __restrict__ whh,  const float* __restrict__ bih,
                            const float* __restrict__ bhh,
                            _Float16* __restrict__ wihf, _Float16* __restrict__ whhf,
                            float* __restrict__ biasf) {
  int i = blockIdx.x * blockDim.x + threadIdx.x;
  if (i < 8192)        wihf[i] = (_Float16)wih0[i];
  else if (i < 24576)  wihf[i] = (_Float16)wihr[i - 8192];
  else if (i < 45056)  whhf[i - 24576] = (_Float16)whh[i - 24576];
  else if (i < 45696) { int j = i - 45056; biasf[j] = bih[j] + bhh[j]; }
}

// ---- one LSTM layer over all timesteps --------------------------------------
// grid = 16 blocks (batch tiles of 16 rows), block = 128 threads = 4 waves.
// Wave w owns gate w (PyTorch order i,f,g,o) = columns [32w, 32w+32) of z.
// DIN==64: read x0 [B,T,64] f32.  DIN==32: read seq [T,B,32] f16 (in place).
template <int DIN>
__global__ void __launch_bounds__(128)
lstm_layer_kernel(const float* __restrict__ x0,
                  const _Float16* __restrict__ wih,  // [128, DIN] f16 row-major
                  const _Float16* __restrict__ whh,  // [128, 32]  f16 row-major
                  const float* __restrict__ bias,    // [128]
                  _Float16* __restrict__ seq) {      // [T, B, 32] f16
  __shared__ __align__(16) float    gates[4][16][32];
  __shared__ __align__(16) _Float16 hbuf[16][32];

  const int tid   = threadIdx.x;
  const int lane  = tid & 31;
  const int wv    = tid >> 5;        // gate index 0..3
  const int ncol  = lane & 15;       // N column within 16-wide tile / A row M
  const int khalf = lane >> 4;       // which K-half this lane holds
  const int b0    = blockIdx.x * 16; // batch tile origin

  // Branchless activation: gate 2 (g) is tanh(x); gates 0,1,3 are
  // sigmoid(x) = 0.5*tanh(0.5*x) + 0.5. Uniform per-wave constants.
  const float tsc = (wv == 2) ? 1.0f : 0.5f;
  const float tsa = (wv == 2) ? 1.0f : 0.5f;
  const float tsb = (wv == 2) ? 0.0f : 0.5f;

  // ---- weights resident in VGPRs for the whole T loop ----
  // B 32x16 layout: lane (ncol,khalf) holds K = 8*khalf..+7 and 16+8*khalf..+7 of column
  // n = wv*32 + nn*16 + ncol  ->  two 16B loads from row-major W (matches A order).
  v16h wihr[DIN / 32][2];  // [kk][nn]
  v16h whhr[2];            // [nn]
  float bv[2];
#pragma unroll
  for (int nn = 0; nn < 2; ++nn) {
    const int ng = wv * 32 + nn * 16 + ncol;
#pragma unroll
    for (int kk = 0; kk < DIN / 32; ++kk)
      wihr[kk][nn] = a_from_f16(wih + (size_t)ng * DIN + kk * 32 + khalf * 8);
    whhr[nn] = a_from_f16(whh + (size_t)ng * HID + khalf * 8);
    bv[nn] = bias[ng];
  }

  // elementwise-update assignment: 4 consecutive columns per thread
  const int um = tid >> 3;        // row 0..15
  const int uc = (tid & 7) * 4;   // col 0,4,...,28
  { h4 z = {(_Float16)0, (_Float16)0, (_Float16)0, (_Float16)0};
    *(h4*)&hbuf[um][uc] = z; }
  float4 c = make_float4(0.f, 0.f, 0.f, 0.f);   // cell state lives in registers
  __syncthreads();

  for (int t = 0; t < T_LEN; ++t) {
    // ---- A operands ----
    v16h ax[DIN / 32], ah;
    if (DIN == 64) {
      const float* xrow = x0 + ((size_t)(b0 + ncol) * T_LEN + t) * 64;
#pragma unroll
      for (int kk = 0; kk < DIN / 32; ++kk)
        ax[kk] = a_from_f32(xrow + kk * 32 + khalf * 8);
      if (t + 1 < T_LEN) __builtin_prefetch(xrow + 64);
    } else {
      const _Float16* xrow = seq + ((size_t)t * B_SZ + b0 + ncol) * HID;
      ax[0] = a_from_f16(xrow + khalf * 8);
      if (t + 1 < T_LEN) __builtin_prefetch(xrow + (size_t)B_SZ * HID);
    }
    ah = a_from_f16(&hbuf[ncol][khalf * 8]);

    // ---- z = x @ Wih^T + h @ Whh^T + b  (this wave's 32 columns) ----
    v8f acc[2];
#pragma unroll
    for (int nn = 0; nn < 2; ++nn) {
#pragma unroll
      for (int j = 0; j < 8; ++j) acc[nn][j] = bv[nn];  // bias in the accumulator
#pragma unroll
      for (int kk = 0; kk < DIN / 32; ++kk)
        acc[nn] = __builtin_amdgcn_wmma_f32_16x16x32_f16(
            false, ax[kk], false, wihr[kk][nn], (short)0, acc[nn], false, false);
      acc[nn] = __builtin_amdgcn_wmma_f32_16x16x32_f16(
          false, ah, false, whhr[nn], (short)0, acc[nn], false, false);
    }

    // ---- activation + publish gate to LDS (C/D layout: M = j + 8*khalf) ----
#pragma unroll
    for (int nn = 0; nn < 2; ++nn)
#pragma unroll
      for (int j = 0; j < 8; ++j) {
        float v = fast_tanh(acc[nn][j] * tsc) * tsa + tsb;
        gates[wv][j + 8 * khalf][nn * 16 + ncol] = v;
      }
    __syncthreads();

    // ---- cell update: c = f*c + i*g ; h = o*tanh(c) ----
    float4 gi = *(const float4*)&gates[0][um][uc];
    float4 gf = *(const float4*)&gates[1][um][uc];
    float4 gg = *(const float4*)&gates[2][um][uc];
    float4 go = *(const float4*)&gates[3][um][uc];
    c.x = gf.x * c.x + gi.x * gg.x;
    c.y = gf.y * c.y + gi.y * gg.y;
    c.z = gf.z * c.z + gi.z * gg.z;
    c.w = gf.w * c.w + gi.w * gg.w;
    h4 hv;
    hv[0] = (_Float16)(go.x * fast_tanh(c.x));
    hv[1] = (_Float16)(go.y * fast_tanh(c.y));
    hv[2] = (_Float16)(go.z * fast_tanh(c.z));
    hv[3] = (_Float16)(go.w * fast_tanh(c.w));
    *(h4*)&hbuf[um][uc] = hv;                                       // next step's A
    *(h4*)(seq + ((size_t)t * B_SZ + b0 + um) * HID + uc) = hv;     // layer output
    __syncthreads();
  }
}

// ---- final FC: out[b,t] = h[t,b,:] . wfc + bfc ------------------------------
__global__ void fc_kernel(const _Float16* __restrict__ seq, const float* __restrict__ wfc,
                          const float* __restrict__ bfc, float* __restrict__ out) {
  int gid = blockIdx.x * blockDim.x + threadIdx.x;
  if (gid >= B_SZ * T_LEN) return;
  int b = gid / T_LEN, t = gid % T_LEN;
  const _Float16* hrow = seq + ((size_t)t * B_SZ + b) * HID;
  float s = bfc[0];
#pragma unroll
  for (int k = 0; k < HID; ++k) s += (float)hrow[k] * wfc[k];
  out[gid] = s;
}

extern "C" void kernel_launch(void* const* d_in, const int* in_sizes, int n_in,
                              void* d_out, int out_size, void* d_ws, size_t ws_size,
                              hipStream_t stream) {
  (void)in_sizes; (void)n_in; (void)out_size; (void)ws_size;
  const float* x    = (const float*)d_in[0];  // [B,T,64]
  const float* Wih0 = (const float*)d_in[1];  // [128,64]
  const float* Wihr = (const float*)d_in[2];  // [4,128,32]
  const float* Whh  = (const float*)d_in[3];  // [5,128,32]
  const float* bih  = (const float*)d_in[4];  // [5,128]
  const float* bhh  = (const float*)d_in[5];  // [5,128]
  const float* Wfc  = (const float*)d_in[6];  // [1,32]
  const float* bfc  = (const float*)d_in[7];  // [1]
  float* out = (float*)d_out;                 // [B,T,1]

  // workspace layout
  char* ws = (char*)d_ws;
  _Float16* wihf  = (_Float16*)ws;                       // 24576 halves
  _Float16* whhf  = wihf + 24576;                        // 20480 halves
  float*    biasf = (float*)(ws + (24576 + 20480) * 2);  // 640 f32
  _Float16* seq   = (_Float16*)(ws + 98304);             // [T,B,32] f16 = 33.5 MB

  prep_kernel<<<(45696 + 255) / 256, 256, 0, stream>>>(Wih0, Wihr, Whh, bih, bhh,
                                                       wihf, whhf, biasf);

  // layer 0: reads f32 x, DIN = 64
  lstm_layer_kernel<64><<<B_SZ / 16, 128, 0, stream>>>(x, wihf, whhf, biasf, seq);
  // layers 1..4: in-place on seq, DIN = 32
  for (int l = 1; l < 5; ++l) {
    lstm_layer_kernel<32><<<B_SZ / 16, 128, 0, stream>>>(
        nullptr, wihf + 8192 + (size_t)(l - 1) * 4096, whhf + (size_t)l * 4096,
        biasf + (size_t)l * G4H, seq);
  }

  fc_kernel<<<(B_SZ * T_LEN + 255) / 256, 256, 0, stream>>>(seq, Wfc, bfc, out);
}
